// TrafficRepresentationNet_2405181685777
// MI455X (gfx1250) — matrix operand order repaired and checked by reading, not compiled, not run on hardware
//
#include <hip/hip_runtime.h>
#include <hip/hip_bf16.h>
#include <cstddef>

// ---------------------------------------------------------------------------
// TrafficRepresentationNet on MI455X (gfx1250, wave32).
// Heavy per-edge MLPs use v_wmma_f32_16x16x32_f16 (f16 in, f32 accumulate).
// Segment max via float atomic-max (int max / uint min trick).
// ---------------------------------------------------------------------------

#define NN 20000
#define NE 320000

typedef __attribute__((ext_vector_type(16))) _Float16 v16h;
typedef __attribute__((ext_vector_type(8)))  float    v8f;

__device__ __forceinline__ v8f wmma_16x16x32(v16h a, v16h b, v8f c) {
  // D = A(16x32 f16) * B(32x16 f16) + C(16x16 f32)
  return __builtin_amdgcn_wmma_f32_16x16x32_f16(false, a, false, b, (short)0, c,
                                                false, false);
}

// A fragment (16x32 f16, row-major source, row stride ldk halves).
// Lane l: row m = l&15; elements 0..7 -> K = k0 + hb + 0..7,
//         elements 8..15 -> K = k0 + hb + 16..23, hb = (l&16)?8:0.
__device__ __forceinline__ v16h load_a_frag(const _Float16* A, int ldk, int k0,
                                            int lane) {
  const _Float16* p = A + (size_t)(lane & 15) * ldk + k0 + ((lane & 16) ? 8 : 0);
  v16h r;
#pragma unroll
  for (int e = 0; e < 8; ++e) { r[e] = p[e]; r[e + 8] = p[e + 16]; }
  return r;
}

// B fragment (32x16 f16). WT is the weight pre-transposed to [Nout][K] f16.
// Lane l: col n (passed in), elements 0..7 -> K = k0 + hb + 0..7,
// elements 8..15 -> K = k0 + hb + 16..23, hb = (l&16)?8:0.
__device__ __forceinline__ v16h load_b_frag(const _Float16* WT, int K, int k0,
                                            int n, int lane) {
  const _Float16* p = WT + (size_t)n * K + k0 + ((lane & 16) ? 8 : 0);
  v16h r;
#pragma unroll
  for (int e = 0; e < 16; ++e) r[e] = p[e < 8 ? e : e + 8];
  return r;
}

// Float atomic max (works with -inf init): sign-aware int max / uint min.
__device__ __forceinline__ void atomicMaxF(float* addr, float v) {
  if (v >= 0.0f) atomicMax((int*)addr, __float_as_int(v));
  else           atomicMin((unsigned int*)addr, __float_as_uint(v));
}

// ---------------------------------------------------------------------------
// Small utility kernels
// ---------------------------------------------------------------------------
__global__ void fill_kernel(float* p, float v, int n) {
  int i = blockIdx.x * blockDim.x + threadIdx.x;
  if (i < n) p[i] = v;
}

__global__ void fixneg_kernel(float* p, int n) {
  int i = blockIdx.x * blockDim.x + threadIdx.x;
  if (i < n) { float v = p[i]; if (v < -3.0e38f) p[i] = 0.0f; }
}

// w [K][NO] f32 row-major  ->  wt [NO][K] f16
__global__ void wtrans_kernel(const float* __restrict__ w,
                              _Float16* __restrict__ wt, int K, int NO) {
  int i = blockIdx.x * blockDim.x + threadIdx.x;
  if (i >= K * NO) return;
  int k = i / NO, n = i % NO;
  wt[(size_t)n * K + k] = (_Float16)w[i];
}

// node_emb = relu((x*0.01) @ W[2,16] + b) -> f16 [N,16]
__global__ void node_emb_kernel(const float* __restrict__ x,
                                const float* __restrict__ w,
                                const float* __restrict__ b,
                                _Float16* __restrict__ out, int n) {
  int i = blockIdx.x * blockDim.x + threadIdx.x;
  if (i >= n) return;
  float x0 = x[i * 2 + 0] * 0.01f, x1 = x[i * 2 + 1] * 0.01f;
#pragma unroll
  for (int o = 0; o < 16; ++o) {
    float v = x0 * w[o] + x1 * w[16 + o] + b[o];
    out[i * 16 + o] = (_Float16)(v > 0.f ? v : 0.f);
  }
}

// edge_emb = relu((edge_attr[:,5:8]*scale) @ W[3,64] + b) -> f16 [E,64]
__global__ void edge_emb_kernel(const float* __restrict__ ea,
                                const float* __restrict__ w,
                                const float* __restrict__ b,
                                _Float16* __restrict__ out, int ne) {
  __shared__ float wl[3 * 64 + 64];
  for (int t = threadIdx.x; t < 256; t += blockDim.x)
    wl[t] = (t < 192) ? w[t] : b[t - 192];
  __syncthreads();
  int e = blockIdx.x * blockDim.x + threadIdx.x;
  if (e >= ne) return;
  float a0 = ea[e * 8 + 5] * 0.01f;
  float a1 = ea[e * 8 + 6];
  float a2 = ea[e * 8 + 7];
#pragma unroll 8
  for (int o = 0; o < 64; ++o) {
    float v = a0 * wl[o] + a1 * wl[64 + o] + a2 * wl[128 + o] + wl[192 + o];
    out[(size_t)e * 64 + o] = (_Float16)(v > 0.f ? v : 0.f);
  }
}

// BatchNorm training-mode stats (biased var), one block per channel.
__global__ void bn_stats_kernel(const float* __restrict__ x, int n, int C,
                                float* __restrict__ mu, float* __restrict__ rs) {
  int c = blockIdx.x;
  float s = 0.f, s2 = 0.f;
  for (int i = threadIdx.x; i < n; i += blockDim.x) {
    float v = x[(size_t)i * C + c];
    s += v; s2 += v * v;
  }
  __shared__ float r1[256], r2[256];
  r1[threadIdx.x] = s; r2[threadIdx.x] = s2;
  __syncthreads();
  for (int st = 128; st > 0; st >>= 1) {
    if ((int)threadIdx.x < st) {
      r1[threadIdx.x] += r1[threadIdx.x + st];
      r2[threadIdx.x] += r2[threadIdx.x + st];
    }
    __syncthreads();
  }
  if (threadIdx.x == 0) {
    float m = r1[0] / n;
    float var = r2[0] / n - m * m;
    mu[c] = m;
    rs[c] = rsqrtf(var + 1e-5f);
  }
}

// y = relu((x-mu)*rs*g + b); optional f32 and f16 outputs.
__global__ void bn_apply_kernel(const float* __restrict__ x,
                                const float* __restrict__ mu,
                                const float* __restrict__ rs,
                                const float* __restrict__ g,
                                const float* __restrict__ b,
                                float* __restrict__ out_f,
                                _Float16* __restrict__ out_h, int total, int C) {
  int i = blockIdx.x * blockDim.x + threadIdx.x;
  if (i >= total) return;
  int c = i % C;
  float v = (x[i] - mu[c]) * rs[c] * g[c] + b[c];
  v = v > 0.f ? v : 0.f;
  if (out_f) out_f[i] = v;
  if (out_h) out_h[i] = (_Float16)v;
}

// ---------------------------------------------------------------------------
// EdgeConv: h = relu([x_i, x_j-x_i] @ W1 + b1) @ W2 + b2; atomic seg-max(dst)
// One wave per 16-edge tile; WMMA for both layers; LDS reshapes L1 -> L2 A.
// ---------------------------------------------------------------------------
template <int CI, int CO>
__global__ void __launch_bounds__(128)
edgeconv_kernel(const _Float16* __restrict__ xh,   // [N,CI] f16
                const int* __restrict__ ei,        // [2,E]
                const _Float16* __restrict__ w1t,  // [CO][2CI] f16
                const float* __restrict__ b1,
                const _Float16* __restrict__ w2t,  // [CO][CO] f16
                const float* __restrict__ b2,
                float* __restrict__ outmax,        // [N,CO] pre-filled -inf
                int ne) {
  constexpr int K1 = 2 * CI;
  constexpr int WPB = 4;
  __shared__ _Float16 a_all[WPB * 16 * K1];
  __shared__ _Float16 h_all[WPB * 16 * CO];
  const int lane = threadIdx.x & 31;
  const int wid  = threadIdx.x >> 5;
  const int tile = blockIdx.x * WPB + wid;   // grid sized exactly, E%64==0
  const int e0 = tile * 16;
  _Float16* a_lds = a_all + wid * 16 * K1;
  _Float16* h_lds = h_all + wid * 16 * CO;

  if (lane == 0) {  // CDNA5 speculative prefetch of next tile's indices
    __builtin_prefetch(ei + e0 + 16, 0, 0);
    __builtin_prefetch(ei + ne + e0 + 16, 0, 0);
  }

  // Stage A rows = [x_i, x_j - x_i]
  for (int idx = lane; idx < 16 * K1; idx += 32) {
    int row = idx / K1, k = idx % K1;
    int e = e0 + row;
    int s = ei[e], d = ei[ne + e];
    float v;
    if (k < CI) v = (float)xh[(size_t)d * CI + k];
    else        v = (float)xh[(size_t)s * CI + (k - CI)] -
                    (float)xh[(size_t)d * CI + (k - CI)];
    a_lds[row * K1 + k] = (_Float16)v;
  }
  __syncthreads();

  v16h af[K1 / 32];
#pragma unroll
  for (int kf = 0; kf < K1 / 32; ++kf)
    af[kf] = load_a_frag(a_lds, K1, kf * 32, lane);

  const int nlo = lane & 15;
  const int mb  = (lane & 16) ? 8 : 0;

  // Layer 1 + ReLU -> LDS (C layout: lane=n, vgpr=m)
#pragma unroll
  for (int nt = 0; nt < CO / 16; ++nt) {
    int n = nt * 16 + nlo;
    float bv = b1[n];
    v8f acc;
#pragma unroll
    for (int i = 0; i < 8; ++i) acc[i] = bv;
#pragma unroll
    for (int kf = 0; kf < K1 / 32; ++kf)
      acc = wmma_16x16x32(af[kf], load_b_frag(w1t, K1, kf * 32, n, lane), acc);
#pragma unroll
    for (int r = 0; r < 8; ++r) {
      float v = acc[r];
      h_lds[(mb + r) * CO + n] = (_Float16)(v > 0.f ? v : 0.f);
    }
  }
  __syncthreads();

  v16h a2[CO / 32];
#pragma unroll
  for (int kf = 0; kf < CO / 32; ++kf)
    a2[kf] = load_a_frag(h_lds, CO, kf * 32, lane);

  int drow[8];
#pragma unroll
  for (int r = 0; r < 8; ++r) drow[r] = ei[ne + e0 + mb + r];

  // Layer 2 + atomic segment max
#pragma unroll
  for (int nt = 0; nt < CO / 16; ++nt) {
    int n = nt * 16 + nlo;
    float bv = b2[n];
    v8f acc;
#pragma unroll
    for (int i = 0; i < 8; ++i) acc[i] = bv;
#pragma unroll
    for (int kf = 0; kf < CO / 32; ++kf)
      acc = wmma_16x16x32(a2[kf], load_b_frag(w2t, CO, kf * 32, n, lane), acc);
#pragma unroll
    for (int r = 0; r < 8; ++r)
      atomicMaxF(&outmax[(size_t)drow[r] * CO + n], acc[r]);
  }
}

// ---------------------------------------------------------------------------
// Generic dense row GEMM: out = act(A[rows,K] @ W[K,NO] + bias)
// ACT: 0=none, 1=relu, 2=leaky(0.2). One wave per 16 rows.
// ---------------------------------------------------------------------------
template <int K, int NO, int ACT>
__global__ void __launch_bounds__(128)
gemm_rows_kernel(const _Float16* __restrict__ Ah,  // [rows,K] f16
                 const _Float16* __restrict__ WT,  // [NO][K] f16
                 const float* __restrict__ bias,   // [NO] or null
                 float* __restrict__ out_f,        // [rows,NO] or null
                 _Float16* __restrict__ out_h,     // [rows,NO] or null
                 int rows) {
  const int lane = threadIdx.x & 31;
  const int tile = (blockIdx.x * blockDim.x + threadIdx.x) >> 5;
  if (tile * 16 >= rows) return;
  const size_t r0 = (size_t)tile * 16;

  v16h af[K / 32];
#pragma unroll
  for (int kf = 0; kf < K / 32; ++kf)
    af[kf] = load_a_frag(Ah + r0 * K, K, kf * 32, lane);

  const int nlo = lane & 15;
  const int mb  = (lane & 16) ? 8 : 0;
#pragma unroll
  for (int nt = 0; nt < NO / 16; ++nt) {
    int n = nt * 16 + nlo;
    float bv = bias ? bias[n] : 0.f;
    v8f acc;
#pragma unroll
    for (int i = 0; i < 8; ++i) acc[i] = bv;
#pragma unroll
    for (int kf = 0; kf < K / 32; ++kf)
      acc = wmma_16x16x32(af[kf], load_b_frag(WT, K, kf * 32, n, lane), acc);
#pragma unroll
    for (int r = 0; r < 8; ++r) {
      float v = acc[r];
      if (ACT == 1) v = v > 0.f ? v : 0.f;
      if (ACT == 2) v = v > 0.f ? v : 0.2f * v;
      size_t o = (r0 + mb + r) * NO + n;
      if (out_f) out_f[o] = v;
      if (out_h) out_h[o] = (_Float16)v;
    }
  }
}

// ---------------------------------------------------------------------------
// EGAT attention: alpha = leaky([x_i, x_j, ea] @ att_w[256,3]); seg-max -> m
// ---------------------------------------------------------------------------
__global__ void egat_alpha_kernel(const _Float16* __restrict__ xlh,  // [N,64]
                                  const _Float16* __restrict__ eah,  // [E,128]
                                  const int* __restrict__ ei,
                                  const float* __restrict__ attw,   // [256,3]
                                  float* __restrict__ alpha,        // [E,3]
                                  float* __restrict__ mbuf,         // [N,3] -inf
                                  int ne) {
  __shared__ float wl[256 * 3];
  for (int t = threadIdx.x; t < 768; t += blockDim.x) wl[t] = attw[t];
  __syncthreads();
  int e = blockIdx.x * blockDim.x + threadIdx.x;
  if (e >= ne) return;
  int s = ei[e], d = ei[ne + e];
  const _Float16* xi = xlh + (size_t)d * 64;
  const _Float16* xj = xlh + (size_t)s * 64;
  const _Float16* ep = eah + (size_t)e * 128;
  float a0 = 0.f, a1 = 0.f, a2 = 0.f;
#pragma unroll 8
  for (int k = 0; k < 64; ++k) {
    float f = (float)xi[k];
    a0 += f * wl[k * 3 + 0]; a1 += f * wl[k * 3 + 1]; a2 += f * wl[k * 3 + 2];
  }
#pragma unroll 8
  for (int k = 0; k < 64; ++k) {
    float f = (float)xj[k]; int kk = 64 + k;
    a0 += f * wl[kk * 3 + 0]; a1 += f * wl[kk * 3 + 1]; a2 += f * wl[kk * 3 + 2];
  }
#pragma unroll 8
  for (int k = 0; k < 128; ++k) {
    float f = (float)ep[k]; int kk = 128 + k;
    a0 += f * wl[kk * 3 + 0]; a1 += f * wl[kk * 3 + 1]; a2 += f * wl[kk * 3 + 2];
  }
  a0 = a0 > 0.f ? a0 : 0.2f * a0;
  a1 = a1 > 0.f ? a1 : 0.2f * a1;
  a2 = a2 > 0.f ? a2 : 0.2f * a2;
  alpha[e * 3 + 0] = a0; alpha[e * 3 + 1] = a1; alpha[e * 3 + 2] = a2;
  atomicMaxF(&mbuf[d * 3 + 0], a0);
  atomicMaxF(&mbuf[d * 3 + 1], a1);
  atomicMaxF(&mbuf[d * 3 + 2], a2);
}

// a = exp(alpha - m[dst]); seg-sum -> s
__global__ void egat_expsum_kernel(const float* __restrict__ alpha,
                                   const float* __restrict__ mbuf,
                                   const int* __restrict__ ei,
                                   float* __restrict__ aexp,
                                   float* __restrict__ sbuf, int ne) {
  int e = blockIdx.x * blockDim.x + threadIdx.x;
  if (e >= ne) return;
  int d = ei[ne + e];
#pragma unroll
  for (int h = 0; h < 3; ++h) {
    float av = expf(alpha[e * 3 + h] - mbuf[d * 3 + h]);
    aexp[e * 3 + h] = av;
    atomicAdd(&sbuf[d * 3 + h], av);
  }
}

// msg = ([x_j, ea] @ lin_w[192,64] + lin_b) * a_norm[e,h]; seg-max -> agg
__global__ void __launch_bounds__(128)
egat_msg_kernel(const _Float16* __restrict__ xlh,    // [N,64]
                const _Float16* __restrict__ eah,    // [E,128]
                const int* __restrict__ ei,
                const _Float16* __restrict__ linwt,  // [64][192]
                const float* __restrict__ linb,
                const float* __restrict__ aexp,      // [E,3]
                const float* __restrict__ sbuf,      // [N,3]
                float* __restrict__ agg,             // [N,3,64] -inf
                int ne) {
  constexpr int K = 192;
  constexpr int WPB = 4;
  __shared__ _Float16 a_all[WPB * 16 * K];
  const int lane = threadIdx.x & 31;
  const int wid  = threadIdx.x >> 5;
  const int tile = blockIdx.x * WPB + wid;
  const int e0 = tile * 16;
  _Float16* a_lds = a_all + wid * 16 * K;

  for (int idx = lane; idx < 16 * K; idx += 32) {
    int row = idx / K, k = idx % K;
    int e = e0 + row;
    _Float16 v;
    if (k < 64) v = xlh[(size_t)ei[e] * 64 + k];
    else        v = eah[(size_t)e * 128 + (k - 64)];
    a_lds[row * K + k] = v;
  }
  __syncthreads();

  v16h af[K / 32];
#pragma unroll
  for (int kf = 0; kf < K / 32; ++kf)
    af[kf] = load_a_frag(a_lds, K, kf * 32, lane);

  const int nlo = lane & 15;
  const int mb  = (lane & 16) ? 8 : 0;
  int drow[8];
  float an[8][3];
#pragma unroll
  for (int r = 0; r < 8; ++r) {
    int e = e0 + mb + r;
    int d = ei[ne + e];
    drow[r] = d;
#pragma unroll
    for (int h = 0; h < 3; ++h)
      an[r][h] = aexp[e * 3 + h] / (sbuf[d * 3 + h] + 1e-16f);
  }

#pragma unroll
  for (int nt = 0; nt < 4; ++nt) {
    int n = nt * 16 + nlo;
    float bv = linb[n];
    v8f acc;
#pragma unroll
    for (int i = 0; i < 8; ++i) acc[i] = bv;
#pragma unroll
    for (int kf = 0; kf < K / 32; ++kf)
      acc = wmma_16x16x32(af[kf], load_b_frag(linwt, K, kf * 32, n, lane), acc);
#pragma unroll
    for (int r = 0; r < 8; ++r) {
      float base = acc[r];
#pragma unroll
      for (int h = 0; h < 3; ++h)
        atomicMaxF(&agg[((size_t)drow[r] * 3 + h) * 64 + n], base * an[r][h]);
    }
  }
}

// pre[n, h*64+c] = where(agg==-inf, 0, agg) + xl[n,c]
__global__ void egat_combine_kernel(const float* __restrict__ agg,
                                    const float* __restrict__ xlf,
                                    float* __restrict__ pre, int n) {
  int i = blockIdx.x * blockDim.x + threadIdx.x;
  if (i >= n * 192) return;
  int node = i / 192, cc = i % 64;
  float v = agg[i];
  if (v < -3.0e38f) v = 0.f;
  pre[i] = v + xlf[node * 64 + cc];
}

// dec = sigmoid(emb @ W[64,8] + b)
__global__ void dec_kernel(const float* __restrict__ emb,
                           const float* __restrict__ w,
                           const float* __restrict__ b,
                           float* __restrict__ out, int n) {
  __shared__ float wl[64 * 8 + 8];
  for (int t = threadIdx.x; t < 520; t += blockDim.x)
    wl[t] = (t < 512) ? w[t] : b[t - 512];
  __syncthreads();
  int i = blockIdx.x * blockDim.x + threadIdx.x;
  if (i >= n) return;
  const float* ep = emb + (size_t)i * 64;
  float acc[8];
#pragma unroll
  for (int o = 0; o < 8; ++o) acc[o] = wl[512 + o];
  for (int k = 0; k < 64; ++k) {
    float f = ep[k];
#pragma unroll
    for (int o = 0; o < 8; ++o) acc[o] += f * wl[k * 8 + o];
  }
#pragma unroll
  for (int o = 0; o < 8; ++o) out[i * 8 + o] = 1.f / (1.f + expf(-acc[o]));
}

// ---------------------------------------------------------------------------
// Host
// ---------------------------------------------------------------------------
static inline int ceil_div(int a, int b) { return (a + b - 1) / b; }

extern "C" void kernel_launch(void* const* d_in, const int* in_sizes, int n_in,
                              void* d_out, int out_size, void* d_ws,
                              size_t ws_size, hipStream_t stream) {
  (void)in_sizes; (void)n_in; (void)out_size; (void)ws_size;
  const int N = NN, E = NE;

  // Input order: top-level dict insertion order (x, edge_index, edge_attr),
  // then params flattened jax-tree style (keys sorted recursively).
  const float* x     = (const float*)d_in[0];
  const int*   ei    = (const int*)  d_in[1];   // [2,E]
  const float* eattr = (const float*)d_in[2];   // [E,8]
  const float* bn1_b  = (const float*)d_in[3];
  const float* bn1_g  = (const float*)d_in[4];
  const float* bn12_b = (const float*)d_in[5];
  const float* bn12_g = (const float*)d_in[6];
  const float* bn2_b  = (const float*)d_in[7];
  const float* bn2_g  = (const float*)d_in[8];
  const float* bn22_b = (const float*)d_in[9];
  const float* bn22_g = (const float*)d_in[10];
  const float* c1_b1 = (const float*)d_in[11];
  const float* c1_b2 = (const float*)d_in[12];
  const float* c1_w1 = (const float*)d_in[13];  // [32,64]
  const float* c1_w2 = (const float*)d_in[14];  // [64,64]
  const float* c2_b1 = (const float*)d_in[15];
  const float* c2_b2 = (const float*)d_in[16];
  const float* c2_w1 = (const float*)d_in[17];  // [128,128]
  const float* c2_w2 = (const float*)d_in[18];  // [128,128]
  // conv3 / conv4: att_w, emb_w, lin_b, lin_w, lin_x_b, lin_x_w
  const float* g3_attw  = (const float*)d_in[19];  // [256,3]
  const float* g3_embw  = (const float*)d_in[20];  // [64,128]
  const float* g3_linb  = (const float*)d_in[21];
  const float* g3_linw  = (const float*)d_in[22];  // [192,64]
  const float* g3_linxb = (const float*)d_in[23];
  const float* g3_linxw = (const float*)d_in[24];  // [128,64]
  const float* g4_attw  = (const float*)d_in[25];
  const float* g4_embw  = (const float*)d_in[26];
  const float* g4_linb  = (const float*)d_in[27];
  const float* g4_linw  = (const float*)d_in[28];
  const float* g4_linxb = (const float*)d_in[29];
  const float* g4_linxw = (const float*)d_in[30];
  const float* dec_b   = (const float*)d_in[31];
  const float* dec_w   = (const float*)d_in[32];  // [64,8]
  const float* medge_b = (const float*)d_in[33];
  const float* medge_w = (const float*)d_in[34];  // [3,64]
  const float* memb_b  = (const float*)d_in[35];
  const float* memb_w  = (const float*)d_in[36];  // [192,64]
  const float* mnode_b = (const float*)d_in[37];
  const float* mnode_w = (const float*)d_in[38];  // [2,16]

  float* out = (float*)d_out;
  float* out_cls  = out;                       // [N,8]
  float* out_reg  = out + (size_t)N * 8;       // [N,8]
  float* out_ccls = out + (size_t)N * 16;      // [N,192]
  float* out_creg = out + (size_t)N * 16 + (size_t)N * 192;

  // Workspace carve-out
  char* wp = (char*)d_ws;
  auto take = [&](size_t bytes) {
    char* q = wp;
    wp += (bytes + 255) & ~(size_t)255;
    return (void*)q;
  };
  _Float16* nemb_h = (_Float16*)take((size_t)N * 16 * 2);
  _Float16* eemb_h = (_Float16*)take((size_t)E * 64 * 2);
  float*    c1raw  = (float*)   take((size_t)N * 64 * 4);
  _Float16* c1n_h  = (_Float16*)take((size_t)N * 64 * 2);
  float*    c2raw  = (float*)   take((size_t)N * 128 * 4);
  _Float16* c2n_h  = (_Float16*)take((size_t)N * 128 * 2);
  float*    xl_f   = (float*)   take((size_t)N * 64 * 4);
  _Float16* xl_h   = (_Float16*)take((size_t)N * 64 * 2);
  _Float16* ea_h   = (_Float16*)take((size_t)E * 128 * 2);
  float*    alpha  = (float*)   take((size_t)E * 3 * 4);
  float*    aexp   = (float*)   take((size_t)E * 3 * 4);
  float*    mbuf   = (float*)   take((size_t)N * 3 * 4);
  float*    sbuf   = (float*)   take((size_t)N * 3 * 4);
  float*    agg    = (float*)   take((size_t)N * 192 * 4);
  float*    pre    = (float*)   take((size_t)N * 192 * 4);
  _Float16* conv_h = (_Float16*)take((size_t)N * 192 * 2);
  float*    emb_f  = (float*)   take((size_t)N * 64 * 4);
  float*    mu     = (float*)   take(192 * 4);
  float*    rs     = (float*)   take(192 * 4);
  _Float16* wt_c1w1 = (_Float16*)take(64 * 32 * 2);
  _Float16* wt_c1w2 = (_Float16*)take(64 * 64 * 2);
  _Float16* wt_c2w1 = (_Float16*)take(128 * 128 * 2);
  _Float16* wt_c2w2 = (_Float16*)take(128 * 128 * 2);
  _Float16* wt_lx3  = (_Float16*)take(64 * 128 * 2);
  _Float16* wt_em3  = (_Float16*)take(128 * 64 * 2);
  _Float16* wt_ln3  = (_Float16*)take(64 * 192 * 2);
  _Float16* wt_lx4  = (_Float16*)take(64 * 128 * 2);
  _Float16* wt_em4  = (_Float16*)take(128 * 64 * 2);
  _Float16* wt_ln4  = (_Float16*)take(64 * 192 * 2);
  _Float16* wt_memb = (_Float16*)take(64 * 192 * 2);

  const int TB = 256;
  auto wtrans = [&](const float* w, _Float16* wt, int K, int NO) {
    wtrans_kernel<<<ceil_div(K * NO, TB), TB, 0, stream>>>(w, wt, K, NO);
  };
  auto fill = [&](float* p, float v, int n) {
    fill_kernel<<<ceil_div(n, TB), TB, 0, stream>>>(p, v, n);
  };

  // --- weight prep (f32 -> transposed f16) ---
  wtrans(c1_w1, wt_c1w1, 32, 64);
  wtrans(c1_w2, wt_c1w2, 64, 64);
  wtrans(c2_w1, wt_c2w1, 128, 128);
  wtrans(c2_w2, wt_c2w2, 128, 128);
  wtrans(g3_linxw, wt_lx3, 128, 64);
  wtrans(g3_embw,  wt_em3, 64, 128);
  wtrans(g3_linw,  wt_ln3, 192, 64);
  wtrans(g4_linxw, wt_lx4, 128, 64);
  wtrans(g4_embw,  wt_em4, 64, 128);
  wtrans(g4_linw,  wt_ln4, 192, 64);
  wtrans(memb_w,   wt_memb, 192, 64);

  // --- feature embeddings ---
  node_emb_kernel<<<ceil_div(N, TB), TB, 0, stream>>>(x, mnode_w, mnode_b,
                                                      nemb_h, N);
  edge_emb_kernel<<<ceil_div(E, TB), TB, 0, stream>>>(eattr, medge_w, medge_b,
                                                      eemb_h, E);

  const int edge_blocks = E / 64;  // 4 waves/block * 16 edges/wave

  // --- conv1: EdgeConv 16 -> 64 ---
  fill(c1raw, -INFINITY, N * 64);
  edgeconv_kernel<16, 64><<<edge_blocks, 128, 0, stream>>>(
      nemb_h, ei, wt_c1w1, c1_b1, wt_c1w2, c1_b2, c1raw, E);
  fixneg_kernel<<<ceil_div(N * 64, TB), TB, 0, stream>>>(c1raw, N * 64);
  bn_stats_kernel<<<64, 256, 0, stream>>>(c1raw, N, 64, mu, rs);
  bn_apply_kernel<<<ceil_div(N * 64, TB), TB, 0, stream>>>(
      c1raw, mu, rs, bn1_g, bn1_b, (float*)nullptr, c1n_h, N * 64, 64);

  // --- conv2: EdgeConv 64 -> 128 ---
  fill(c2raw, -INFINITY, N * 128);
  edgeconv_kernel<64, 128><<<edge_blocks, 128, 0, stream>>>(
      c1n_h, ei, wt_c2w1, c2_b1, wt_c2w2, c2_b2, c2raw, E);
  fixneg_kernel<<<ceil_div(N * 128, TB), TB, 0, stream>>>(c2raw, N * 128);
  bn_stats_kernel<<<128, 256, 0, stream>>>(c2raw, N, 128, mu, rs);
  bn_apply_kernel<<<ceil_div(N * 128, TB), TB, 0, stream>>>(
      c2raw, mu, rs, bn12_g, bn12_b, (float*)nullptr, c2n_h, N * 128, 128);

  // --- EGAT pass (shared for conv3/conv4) ---
  auto run_egat = [&](const float* attw, const _Float16* embwt,
                      const float* linb, const _Float16* linwt,
                      const float* linxb, const _Float16* linxwt,
                      const float* bng, const float* bnb, float* conv_out,
                      float* head_out) {
    const int node_tiles = N / 16;
    const int node_gemm_blocks = ceil_div(node_tiles * 32, 128);
    // xl = c2 @ lin_x_w + lin_x_b
    gemm_rows_kernel<128, 64, 0><<<node_gemm_blocks, 128, 0, stream>>>(
        c2n_h, linxwt, linxb, xl_f, xl_h, N);
    // ea = leaky(edge_emb @ emb_w)
    gemm_rows_kernel<64, 128, 2><<<ceil_div((E / 16) * 32, 128), 128, 0,
                                   stream>>>(eemb_h, embwt, (const float*)nullptr,
                                             (float*)nullptr, ea_h, E);
    fill(mbuf, -INFINITY, N * 3);
    fill(sbuf, 0.0f, N * 3);
    fill(agg, -INFINITY, N * 192);
    egat_alpha_kernel<<<ceil_div(E, TB), TB, 0, stream>>>(xl_h, ea_h, ei, attw,
                                                          alpha, mbuf, E);
    egat_expsum_kernel<<<ceil_div(E, TB), TB, 0, stream>>>(alpha, mbuf, ei,
                                                           aexp, sbuf, E);
    egat_msg_kernel<<<edge_blocks, 128, 0, stream>>>(xl_h, ea_h, ei, linwt,
                                                     linb, aexp, sbuf, agg, E);
    egat_combine_kernel<<<ceil_div(N * 192, TB), TB, 0, stream>>>(agg, xl_f,
                                                                  pre, N);
    bn_stats_kernel<<<192, 256, 0, stream>>>(pre, N, 192, mu, rs);
    bn_apply_kernel<<<ceil_div(N * 192, TB), TB, 0, stream>>>(
        pre, mu, rs, bng, bnb, conv_out, conv_h, N * 192, 192);
    // emb = relu(conv @ mlp_emb_w + b); head = sigmoid(emb @ mlp_dec_w + b)
    gemm_rows_kernel<192, 64, 1><<<node_gemm_blocks, 128, 0, stream>>>(
        conv_h, wt_memb, memb_b, emb_f, (_Float16*)nullptr, N);
    dec_kernel<<<ceil_div(N, TB), TB, 0, stream>>>(emb_f, dec_w, dec_b,
                                                   head_out, N);
  };

  run_egat(g3_attw, wt_em3, g3_linb, wt_ln3, g3_linxb, wt_lx3, bn2_g, bn2_b,
           out_ccls, out_cls);
  run_egat(g4_attw, wt_em4, g4_linb, wt_ln4, g4_linxb, wt_lx4, bn22_g, bn22_b,
           out_creg, out_reg);
}
